// MacroScore_40845138985487
// MI455X (gfx1250) — matrix operand order repaired and checked by reading, not compiled, not run on hardware
//
#include <hip/hip_runtime.h>
#include <hip/hip_bf16.h>

typedef __attribute__((ext_vector_type(16))) _Float16 v16h;
typedef __attribute__((ext_vector_type(8)))  float    v8f;

#define CLASS_NUM 6

// Reconstruct a 3-bit class id for wave slot k from three ballot masks.
__device__ __forceinline__ unsigned decode3(unsigned b0, unsigned b1, unsigned b2, unsigned k) {
    return ((b0 >> k) & 1u) | (((b1 >> k) & 1u) << 1) | (((b2 >> k) & 1u) << 2);
}

__global__ void __launch_bounds__(64)
cm_zero_kernel(float* __restrict__ cm) {
    if (threadIdx.x < 36) cm[threadIdx.x] = 0.0f;
}

// One-hot confusion-matrix accumulation via v_wmma_f32_16x16x32_f16.
// CM[pred, tgt] += 1 over 32 rows per WMMA, accumulator lives in v8f registers.
__global__ void __launch_bounds__(256)
cm_wmma_kernel(const float* __restrict__ x, const int* __restrict__ tgt,
               const float* __restrict__ w, float* __restrict__ cm, long long n)
{
    __shared__ float scm[36];
    if (threadIdx.x < 36) scm[threadIdx.x] = 0.0f;
    __syncthreads();

    const unsigned lane = threadIdx.x & 31u;
    const unsigned mrow = lane & 15u;   // M index (A) / N index (B) / col (C,D)
    const unsigned hf   = lane >> 4;    // half-wave selector
    const unsigned wavesPerBlock = blockDim.x >> 5;
    const long long totalWaves = (long long)gridDim.x * wavesPerBlock;
    const long long myWave = (long long)blockIdx.x * wavesPerBlock + (threadIdx.x >> 5);

    const float w0 = w[0], w1 = w[1], w2 = w[2], w3 = w[3], w4 = w[4], w5 = w[5];

    v8f acc = {};  // 16x16 f32 partial confusion matrix, striped per ISA layout

    const long long nChunks = (n + 31) >> 5;
    for (long long c = myWave; c < nChunks; c += totalWaves) {
        const long long row = (c << 5) + (long long)lane;
        unsigned pred = 15u, tg = 15u;   // OOB rows land at CM[7,7] (ignored)
        if (row < n) {
            const float2* pr = reinterpret_cast<const float2*>(x + row * 6);
            const float2 v01 = pr[0], v23 = pr[1], v45 = pr[2];
            float best = v01.x * w0; pred = 0u;
            float v;
            v = v01.y * w1; if (v > best) { best = v; pred = 1u; }
            v = v23.x * w2; if (v > best) { best = v; pred = 2u; }
            v = v23.y * w3; if (v > best) { best = v; pred = 3u; }
            v = v45.x * w4; if (v > best) { best = v; pred = 4u; }
            v = v45.y * w5; if (v > best) { best = v; pred = 5u; }
            tg = (unsigned)tgt[row] & 15u;
        }
        // Broadcast all 32 preds/targets to every lane: 3 ballots each (3-bit ids).
        const unsigned pb0 = __builtin_amdgcn_ballot_w32((pred & 1u) != 0u);
        const unsigned pb1 = __builtin_amdgcn_ballot_w32((pred & 2u) != 0u);
        const unsigned pb2 = __builtin_amdgcn_ballot_w32((pred & 4u) != 0u);
        const unsigned tb0 = __builtin_amdgcn_ballot_w32((tg   & 1u) != 0u);
        const unsigned tb1 = __builtin_amdgcn_ballot_w32((tg   & 2u) != 0u);
        const unsigned tb2 = __builtin_amdgcn_ballot_w32((tg   & 4u) != 0u);

        // Build one-hot A (16x32, M=pred-class, K=row slot) and B (32x16, N=tgt-class)
        // in the documented f16 WMMA VGPR layouts.
        v16h A, B;
#pragma unroll
        for (int vv = 0; vv < 8; ++vv) {
            // A: lanes 0-15 -> VGPR0-3: K=0..7, VGPR4-7: K=16..23; lanes 16-31: +8
            const unsigned ka = ((unsigned)(vv >> 2) << 4) | (hf << 3) | ((unsigned)(vv & 3) << 1);
            A[2 * vv]     = (decode3(pb0, pb1, pb2, ka)     == mrow) ? (_Float16)1.0f : (_Float16)0.0f;
            A[2 * vv + 1] = (decode3(pb0, pb1, pb2, ka + 1) == mrow) ? (_Float16)1.0f : (_Float16)0.0f;
            // B: lanes 0-15 -> VGPR0-3: K=0..7, VGPR4-7: K=16..23; lanes 16-31: +8
            const unsigned kb = ((unsigned)(vv >> 2) << 4) | (hf << 3) | ((unsigned)(vv & 3) << 1);
            B[2 * vv]     = (decode3(tb0, tb1, tb2, kb)     == mrow) ? (_Float16)1.0f : (_Float16)0.0f;
            B[2 * vv + 1] = (decode3(tb0, tb1, tb2, kb + 1) == mrow) ? (_Float16)1.0f : (_Float16)0.0f;
        }
        acc = __builtin_amdgcn_wmma_f32_16x16x32_f16(
            /*neg_a=*/false, A, /*neg_b=*/false, B,
            /*c_mod=*/(short)0, acc, /*reuse_a=*/false, /*reuse_b=*/false);
    }

    // C/D layout: lane (hf==0, n=mrow) VGPR r holds CM[r, mrow]; rows 6..15 are not read.
    if (hf == 0u && mrow < CLASS_NUM) {
#pragma unroll
        for (int r = 0; r < CLASS_NUM; ++r) {
            const float vv = acc[r];
            if (vv != 0.0f) atomicAdd(&scm[r * CLASS_NUM + (int)mrow], vv); // ds_add_f32
        }
    }
    __syncthreads();
    if (threadIdx.x < 36) {
        const float vv = scm[threadIdx.x];
        if (vv != 0.0f) atomicAdd(&cm[threadIdx.x], vv);  // global_atomic_add_f32
    }
}

__global__ void __launch_bounds__(32)
cm_finalize_kernel(const float* __restrict__ cm, const float* __restrict__ w,
                   float* __restrict__ out)
{
    if (threadIdx.x == 0) {
        float f1sum = 0.0f;
#pragma unroll
        for (int c = 0; c < CLASS_NUM; ++c) {
            const float diag = cm[c * CLASS_NUM + c];
            float colsum = 0.0f, rowsum = 0.0f;
#pragma unroll
            for (int j = 0; j < CLASS_NUM; ++j) {
                colsum += cm[j * CLASS_NUM + c];   // sum over preds  (axis=0)
                rowsum += cm[c * CLASS_NUM + j];   // sum over targets (axis=1)
            }
            const float prec = diag / colsum;
            const float rec  = diag / rowsum;
            f1sum += 2.0f * prec * rec / (prec + rec);
        }
        out[0] = -f1sum / (float)CLASS_NUM;
#pragma unroll
        for (int i = 0; i < CLASS_NUM; ++i) out[1 + i] = w[i];
    }
}

extern "C" void kernel_launch(void* const* d_in, const int* in_sizes, int n_in,
                              void* d_out, int out_size, void* d_ws, size_t ws_size,
                              hipStream_t stream) {
    const float* x   = (const float*)d_in[0];   // inputs   [N,6] f32
    const int*   tgt = (const int*)d_in[1];     // targets  [N,1] i32
    const float* w   = (const float*)d_in[2];   // class_weight [6] f32
    float* out = (float*)d_out;                 // [0]=loss, [1..6]=class_weight
    float* cm  = (float*)d_ws;                  // 36-float confusion matrix

    const long long n = (long long)in_sizes[1]; // N (targets flat count)

    cm_zero_kernel<<<1, 64, 0, stream>>>(cm);
    // 1024 blocks x 256 threads = 8192 waves; N/32 = 262144 WMMA chunks -> 32/wave.
    cm_wmma_kernel<<<1024, 256, 0, stream>>>(x, tgt, w, cm, n);
    cm_finalize_kernel<<<1, 32, 0, stream>>>(cm, w, out);
}
